// LlamaAttention_53961969106970
// MI455X (gfx1250) — compile-verified, matrix operand order checked by reading, not run on hardware
//
#include <hip/hip_runtime.h>

// ---------------------------------------------------------------------------
// LlamaAttention w/ LoRA for MI455X (gfx1250).
// v_wmma_f32_16x16x32_bf16 for all big matmuls; one-time fp32->bf16 converts;
// TDM (tensor_load_to_lds) tile staging with async-to-LDS fallback;
// flash-attention with online softmax.  B=4 S=1024 HID=4096 H=32 HKV=8 D=128 R=16
// ---------------------------------------------------------------------------

#define B_   4
#define S_   1024
#define HID_ 4096
#define H_   32
#define HKV_ 8
#define D_   128
#define R_   16

typedef unsigned short u16;
typedef __attribute__((ext_vector_type(16))) __bf16 v16bf;
typedef __attribute__((ext_vector_type(2)))  __bf16 v2bf;
typedef __attribute__((ext_vector_type(8)))  float  v8f;
typedef __attribute__((ext_vector_type(4)))  unsigned int u32x4;
typedef __attribute__((ext_vector_type(4)))  float  f32x4;

union FragA {
  v16bf v;
  u32x4 x[2];
  u16   u[16];
};

__device__ __forceinline__ u16 f2bf(float f) {
  unsigned u = __builtin_bit_cast(unsigned, f);
  unsigned rb = ((u >> 16) & 1u) + 0x7FFFu;   // round-nearest-even
  return (u16)((u + rb) >> 16);
}
__device__ __forceinline__ float bf2f(u16 h) {
  unsigned u = ((unsigned)h) << 16;
  return __builtin_bit_cast(float, u);
}
__device__ __forceinline__ unsigned cvt_pk_bf16(float a, float b) {
#if __has_builtin(__builtin_amdgcn_cvt_pk_bf16_f32)
  v2bf r = __builtin_amdgcn_cvt_pk_bf16_f32(a, b);
  return __builtin_bit_cast(unsigned, r);
#else
  return (unsigned)f2bf(a) | ((unsigned)f2bf(b) << 16);
#endif
}

// -------------------- async global->LDS (CDNA5) ----------------------------
#if __has_builtin(__builtin_amdgcn_global_load_async_to_lds_b128)
#define HAVE_ASYNC_LDS 1
typedef int i32x4 __attribute__((vector_size(16)));
typedef __attribute__((address_space(1))) i32x4 as1_i32x4;
typedef __attribute__((address_space(3))) i32x4 as3_i32x4;
#else
#define HAVE_ASYNC_LDS 0
#endif

__device__ __forceinline__ void async_cp_b128(u16* lds_dst, const u16* gsrc) {
#if HAVE_ASYNC_LDS
  __builtin_amdgcn_global_load_async_to_lds_b128(
      (as1_i32x4*)gsrc, (as3_i32x4*)lds_dst, 0, 0);
#else
  *(u32x4*)lds_dst = *(const u32x4*)gsrc;
#endif
}
__device__ __forceinline__ void async_wait0() {
#if HAVE_ASYNC_LDS
#if __has_builtin(__builtin_amdgcn_s_wait_asynccnt)
  __builtin_amdgcn_s_wait_asynccnt(0);
#else
  asm volatile("s_wait_asynccnt 0" ::: "memory");
#endif
#endif
}

// -------------------- TDM: tensor_load_to_lds (CDNA5) ----------------------
#if __has_builtin(__builtin_amdgcn_tensor_load_to_lds)
#define HAVE_TDM 1
typedef __attribute__((ext_vector_type(8))) int i32x8;
typedef __attribute__((ext_vector_type(4))) int i32x4s;
typedef __attribute__((address_space(3))) const void as3_cvoid;

__device__ __forceinline__ unsigned lds_offset_of(const void* p) {
  return (unsigned)(unsigned long long)(as3_cvoid*)p;
}

// 2D tile load: d1 rows of d0 bf16 elements, global row stride `stride`
// elements; LDS rows padded by (pad_amt_code+1) DWORDs every
// 2^(pad_int_code+1) DWORDs. D# layout per CDNA5 ISA 8.3/8.4.
__device__ __forceinline__ void tdm_load_2d(
    unsigned lds_addr, const void* gp, int d0, int d1, long stride,
    int pad_int_code, int pad_amt_code)
{
  const unsigned long long ga = (unsigned long long)gp;
  u32x4 g0;
  g0[0] = 1u;                                   // count=1, user mode
  g0[1] = lds_addr;                             // lds_addr
  g0[2] = (unsigned)ga;                         // global_addr[31:0]
  g0[3] = (unsigned)((ga >> 32) & 0x1FFFFFFu)   // global_addr[56:32]
          | (2u << 30);                         // type=2 ("image")
  i32x8 g1;
  g1[0] = (1 << 16)                             // data_size = 2 bytes
        | (1 << 20)                             // pad_enable
        | (pad_int_code << 22)                  // pad interval
        | (pad_amt_code << 25);                 // pad amount
  g1[1] = (d0 & 0xFFFF) << 16;                  // tensor_dim0[15:0]
  g1[2] = ((d1 & 0xFFFF) << 16);                // tensor_dim1[15:0]
  g1[3] = (d0 & 0xFFFF) << 16;                  // tile_dim0
  g1[4] = d1 & 0xFFFF;                          // tile_dim1 (tile_dim2=0)
  g1[5] = (int)(stride & 0xFFFFFFFFll);         // tensor_dim0_stride[31:0]
  g1[6] = (int)((stride >> 32) & 0xFFFFll);     // tensor_dim0_stride[47:32]
  g1[7] = 0;
  i32x4s gz4 = {0, 0, 0, 0};
  i32x8  gz8 = {0, 0, 0, 0, 0, 0, 0, 0};
  __builtin_amdgcn_tensor_load_to_lds(g0, g1, gz4, gz4, gz8, 0);
}
__device__ __forceinline__ void tensor_wait0() {
#if __has_builtin(__builtin_amdgcn_s_wait_tensorcnt)
  __builtin_amdgcn_s_wait_tensorcnt(0);
#else
  asm volatile("s_wait_tensorcnt 0" ::: "memory");
#endif
}
#else
#define HAVE_TDM 0
#endif

// Load one 16x32 bf16 A/B fragment (per documented gfx1250 VGPR layout):
// lanes 0-15 hold K {0..7,16..23}, lanes 16-31 hold K {8..15,24..31}.
__device__ __forceinline__ void frag16(FragA& f, const u16* rowp, int koff) {
  f.x[0] = *(const u32x4*)(rowp + koff);
  f.x[1] = *(const u32x4*)(rowp + 16 + koff);
}

// ---------------------------------------------------------------------------
// One-time fp32 -> bf16 convert (8 elems/thread)
// ---------------------------------------------------------------------------
__global__ __launch_bounds__(256) void cvt_bf16_kernel(
    const float* __restrict__ in, u16* __restrict__ out, size_t n)
{
  const size_t i = ((size_t)blockIdx.x * 256 + threadIdx.x) * 8;
  if (i >= n) return;
  f32x4 a = *(const f32x4*)(in + i);
  f32x4 b = *(const f32x4*)(in + i + 4);
  u32x4 w = { cvt_pk_bf16(a[0], a[1]), cvt_pk_bf16(a[2], a[3]),
              cvt_pk_bf16(b[0], b[1]), cvt_pk_bf16(b[2], b[3]) };
  *(u32x4*)(out + i) = w;
}

// ---------------------------------------------------------------------------
// t[b,s,r] = sum_i x[b,s,i] * A[b,r,i]    (fp32 accumulate, rank 16)
// ---------------------------------------------------------------------------
template<typename InT>
__global__ __launch_bounds__(256) void lora_t_kernel(
    const InT*  __restrict__ x,   // [B*S, K]
    const float* __restrict__ A,  // [B, R, K]
    float* __restrict__ t,        // [B*S, R]
    int K)
{
  const int gid = blockIdx.x * 256 + threadIdx.x;   // B*S*R threads
  const int r = gid & (R_ - 1);
  const size_t row = (size_t)(gid >> 4);
  const int b = (int)(row >> 10);                   // row / S_ (S_=1024)
  const InT* xp = x + row * (size_t)K;
  const float* ap = A + ((size_t)b * R_ + r) * (size_t)K;
  float acc = 0.0f;
  for (int i = 0; i < K; i += 8) {
    float xv[8];
    if constexpr (sizeof(InT) == 2) {
      u32x4 g = *(const u32x4*)(xp + i);
      for (int j = 0; j < 4; ++j) {
        xv[2*j]   = bf2f((u16)(g[j] & 0xffffu));
        xv[2*j+1] = bf2f((u16)(g[j] >> 16));
      }
    } else {
      f32x4 g0 = *(const f32x4*)(xp + i);
      f32x4 g1 = *(const f32x4*)(xp + i + 4);
      xv[0]=g0[0]; xv[1]=g0[1]; xv[2]=g0[2]; xv[3]=g0[3];
      xv[4]=g1[0]; xv[5]=g1[1]; xv[6]=g1[2]; xv[7]=g1[3];
    }
    f32x4 a0 = *(const f32x4*)(ap + i);
    f32x4 a1 = *(const f32x4*)(ap + i + 4);
    acc += xv[0]*a0[0] + xv[1]*a0[1] + xv[2]*a0[2] + xv[3]*a0[3]
         + xv[4]*a1[0] + xv[5]*a1[1] + xv[6]*a1[2] + xv[7]*a1[3];
  }
  t[row * R_ + r] = acc;
}

// ---------------------------------------------------------------------------
// Out[m,n] = sum_k A[m,k]*W[n,k] (+ sum_r t[m,r]*Blora[b,n,r])
// A, W already bf16. 128x128 block tile, 8 waves x (64x32). K-step 32 staged
// through LDS by the TDM (wave 0 issues 2 descriptors) or async copies.
// ---------------------------------------------------------------------------
template<typename OutT>
__global__ __launch_bounds__(256) void gemm_bf16(
    const u16* __restrict__ A,        // [M,K] bf16
    const u16* __restrict__ W,        // [N,K] bf16
    const float* __restrict__ tlora,  // [M,R]
    const float* __restrict__ Blora,  // [B,N,R]
    OutT* __restrict__ Out,           // [M,N]
    int M, int N, int K, int use_lora)
{
  __shared__ __align__(16) u16 As[128][40];   // 64B data + 16B pad per row
  __shared__ __align__(16) u16 Ws[128][40];

  const int tid  = threadIdx.x;
  const int lane = tid & 31;
  const int wave = tid >> 5;
  const int wr = wave >> 2;               // 0..1 : 64-row strip
  const int wc = wave & 3;                // 0..3 : 32-col strip
  const int blockM = blockIdx.y * 128;
  const int blockN = blockIdx.x * 128;
  const int koff = (lane < 16) ? 0 : 8;

  const int ldrow = tid >> 1;             // 0..127
  const int ldkb  = (tid & 1) * 16;       // k-half within the 32-wide step

  v8f zero = 0.0f;
  v8f acc[4][2];
  for (int i = 0; i < 4; ++i)
    for (int j = 0; j < 2; ++j)
      acc[i][j] = zero;

  const u16* agp = A + (size_t)(blockM + ldrow) * (size_t)K + ldkb;
  const u16* wgp = W + (size_t)(blockN + ldrow) * (size_t)K + ldkb;
#if HAVE_TDM
  const u16* atile = A + (size_t)blockM * (size_t)K;   // tile row 0
  const u16* wtile = W + (size_t)blockN * (size_t)K;
  const unsigned lds_as = lds_offset_of(&As[0][0]);
  const unsigned lds_ws = lds_offset_of(&Ws[0][0]);
#endif

  for (int kk = 0; kk < K; kk += 32) {
    __syncthreads();
#if HAVE_TDM
    if (tid < 32) {
      // 128 rows x 32 bf16; LDS pad 4 DWORDs every 16 DWORDs (80B pitch)
      tdm_load_2d(lds_as, atile + kk, 32, 128, (long)K, 3, 3);
      tdm_load_2d(lds_ws, wtile + kk, 32, 128, (long)K, 3, 3);
    }
#else
    async_cp_b128(&As[ldrow][ldkb],     agp + kk);
    async_cp_b128(&As[ldrow][ldkb + 8], agp + kk + 8);
    async_cp_b128(&Ws[ldrow][ldkb],     wgp + kk);
    async_cp_b128(&Ws[ldrow][ldkb + 8], wgp + kk + 8);
#endif
    // prefetch next K-step while WMMAs run on this one
    __builtin_prefetch((const void*)(agp + kk + 32), 0, 0);
    __builtin_prefetch((const void*)(wgp + kk + 32), 0, 0);
#if HAVE_TDM
    if (tid < 32) tensor_wait0();
#else
    async_wait0();
#endif
    __syncthreads();

    FragA af[4], bf[2];
    for (int mt = 0; mt < 4; ++mt)
      frag16(af[mt], &As[wr*64 + mt*16 + (lane & 15)][0], koff);
    for (int nt = 0; nt < 2; ++nt)
      frag16(bf[nt], &Ws[wc*32 + nt*16 + (lane & 15)][0], koff);

    for (int mt = 0; mt < 4; ++mt)
      for (int nt = 0; nt < 2; ++nt)
        acc[mt][nt] = __builtin_amdgcn_wmma_f32_16x16x32_bf16(
            false, af[mt].v, false, bf[nt].v, (short)0, acc[mt][nt],
            false, false);
  }

  // epilogue: LoRA add (fp32) + store
  for (int mt = 0; mt < 4; ++mt) {
    for (int r = 0; r < 8; ++r) {
      const int m = blockM + wr*64 + mt*16 + r + ((lane >> 4) << 3);
      float tvv[R_];
      if (use_lora) {
        const float* tp = tlora + (size_t)m * R_;
        for (int rr = 0; rr < R_; ++rr) tvv[rr] = tp[rr];
      }
      const int batch = m / S_;
      for (int nt = 0; nt < 2; ++nt) {
        const int n = blockN + wc*32 + nt*16 + (lane & 15);
        float val = acc[mt][nt][r];
        if (use_lora) {
          const float* bp = Blora + ((size_t)batch * (size_t)N + n) * R_;
          for (int rr = 0; rr < R_; ++rr) val += tvv[rr] * bp[rr];
        }
        if constexpr (sizeof(OutT) == 2)
          Out[(size_t)m * (size_t)N + n] = (OutT)f2bf(val);
        else
          Out[(size_t)m * (size_t)N + n] = (OutT)val;
      }
    }
  }
}

// ---------------------------------------------------------------------------
// In-place RoPE on bf16 [B,S,nh,D]; scale folds 1/sqrt(D) into q.
// ---------------------------------------------------------------------------
__global__ __launch_bounds__(256) void rope_kernel(
    u16* __restrict__ x, int nheads, float scale, size_t total)
{
  const size_t idx = (size_t)blockIdx.x * 256 + threadIdx.x;
  if (idx >= total) return;
  const int dp = (int)(idx % (D_ / 2));
  size_t t = idx / (D_ / 2);
  const int hh = (int)(t % nheads);
  const size_t bs = t / nheads;
  const int s = (int)(bs % S_);
  const float freq = __powf(10000.0f, -2.0f * (float)dp / (float)D_);
  float c, sn;
  __sincosf((float)s * freq, &sn, &c);
  const size_t base = (bs * (size_t)nheads + hh) * D_;
  const float x1 = bf2f(x[base + dp]);
  const float x2 = bf2f(x[base + dp + D_ / 2]);
  x[base + dp]          = f2bf((x1 * c - x2 * sn) * scale);
  x[base + dp + D_ / 2] = f2bf((x2 * c + x1 * sn) * scale);
}

// ---------------------------------------------------------------------------
// Flash attention: block = (b,h,64-row q tile), 4 waves x 16 q rows.
// K tile (64x128) staged by TDM (or async), V^T tile (128x64) via transpose;
// online softmax. q pre-scaled by 1/sqrt(D) and pre-roped. Output bf16.
// ---------------------------------------------------------------------------
__global__ __launch_bounds__(128) void attn_fa(
    const u16* __restrict__ qb,   // [B,S,H,D] bf16
    const u16* __restrict__ kb,   // [B,S,HKV,D] bf16
    const u16* __restrict__ vb,   // [B,S,HKV,D] bf16
    u16* __restrict__ attn)       // [B,S,H*D] bf16
{
  __shared__ __align__(16) u16 Ks[64][136];     // [kpos][d], 256B + 16B pad
  __shared__ __align__(16) u16 Vt[128][72];     // [d][kpos]
  __shared__ __align__(16) u16 Ps[4][16][72];   // per-wave P scratch

  const int tid  = threadIdx.x;
  const int lane = tid & 31;
  const int wave = tid >> 5;
  const int hi = lane >> 4;
  const int ln = lane & 15;
  const int koff = hi ? 8 : 0;

  int bid = blockIdx.x;
  const int qt = bid & 15;  bid >>= 4;      // S/64 = 16 q tiles
  const int h  = bid & 31;  bid >>= 5;      // H = 32
  const int b  = bid;
  const int kvh = h >> 2;                   // GQA: H/HKV = 4

  FragA qf[4];
  {
    const int qrow = qt * 64 + wave * 16 + ln;
    const u16* qp = qb + (((size_t)(b * S_ + qrow)) * H_ + h) * D_;
    for (int f = 0; f < 4; ++f) frag16(qf[f], qp + 32 * f, koff);
  }

  v8f zero = 0.0f;
  v8f o[8];
  for (int i = 0; i < 8; ++i) o[i] = zero;
  float mrow[8], lrow[8];
  for (int r = 0; r < 8; ++r) { mrow[r] = -1e30f; lrow[r] = 0.0f; }

#if HAVE_TDM
  const unsigned lds_ks = lds_offset_of(&Ks[0][0]);
#endif

  for (int kt = 0; kt <= qt; ++kt) {
    {   // stage K tile (TDM / async) and transposed V tile
      const int krow  = tid >> 1;            // 0..63
      const int dbase = (tid & 1) * 64;
      const size_t kvoff =
          (((size_t)(b * S_ + kt * 64 + krow)) * HKV_ + kvh) * D_ + dbase;
#if HAVE_TDM
      if (tid < 32) {
        const u16* ktile =
            kb + (((size_t)(b * S_ + kt * 64)) * HKV_ + kvh) * D_;
        // 64 rows x 128 bf16; LDS pad 4 DWORDs every 64 DWORDs (272B pitch)
        tdm_load_2d(lds_ks, ktile, 128, 64, (long)(HKV_ * D_), 5, 3);
      }
#else
      const u16* kp = kb + kvoff;
      for (int i = 0; i < 8; ++i)
        async_cp_b128(&Ks[krow][dbase + i * 8], kp + i * 8);
#endif
      const u16* vp = vb + kvoff;
      for (int i = 0; i < 8; ++i) {
        u32x4 g = *(const u32x4*)(vp + i * 8);
        for (int j = 0; j < 4; ++j) {
          unsigned w = g[j];
          Vt[dbase + i * 8 + 2 * j][krow]     = (u16)(w & 0xffffu);
          Vt[dbase + i * 8 + 2 * j + 1][krow] = (u16)(w >> 16);
        }
      }
#if HAVE_TDM
      if (tid < 32) tensor_wait0();
#else
      async_wait0();
#endif
    }
    __syncthreads();

    // S = Q K^T : 16 q rows x 64 keys per wave.
    // Load 4 independent B-fragments per K-step, then 4 independent WMMAs.
    v8f st[4];
    for (int nt = 0; nt < 4; ++nt) st[nt] = zero;
    for (int f = 0; f < 4; ++f) {
      FragA kf[4];
      for (int nt = 0; nt < 4; ++nt)
        frag16(kf[nt], &Ks[nt * 16 + ln][32 * f], koff);
      for (int nt = 0; nt < 4; ++nt)
        st[nt] = __builtin_amdgcn_wmma_f32_16x16x32_bf16(
            false, qf[f].v, false, kf[nt].v, (short)0, st[nt], false, false);
    }

    // causal mask + online softmax (rows in VGPR index, cols in lanes;
    // xor-shuffles 8..1 reduce within each independent 16-lane half)
    const int rowg = qt * 64 + wave * 16 + hi * 8;
    float alpha[8];
    for (int r = 0; r < 8; ++r) {
      const int rg = rowg + r;
      float mx = -1e30f;
      for (int nt = 0; nt < 4; ++nt) {
        const int cg = kt * 64 + nt * 16 + ln;
        float v = st[nt][r];
        if (cg > rg) v = -1e30f;
        st[nt][r] = v;
        mx = fmaxf(mx, v);
      }
      for (int d = 8; d >= 1; d >>= 1)
        mx = fmaxf(mx, __shfl_xor(mx, d, 32));
      const float mn = fmaxf(mrow[r], mx);
      float rs = 0.0f;
      for (int nt = 0; nt < 4; ++nt) {
        float p = __expf(st[nt][r] - mn);
        st[nt][r] = p;
        rs += p;
      }
      for (int d = 8; d >= 1; d >>= 1)
        rs += __shfl_xor(rs, d, 32);
      alpha[r] = __expf(mrow[r] - mn);
      lrow[r] = lrow[r] * alpha[r] + rs;
      mrow[r] = mn;
    }
    for (int nt = 0; nt < 8; ++nt)
      for (int r = 0; r < 8; ++r)
        o[nt][r] *= alpha[r];

    // P (C-layout) -> per-wave LDS -> A-fragment layout
    for (int nt = 0; nt < 4; ++nt)
      for (int r = 0; r < 8; ++r)
        Ps[wave][hi * 8 + r][nt * 16 + ln] = f2bf(st[nt][r]);

    FragA pf[2];
    for (int j = 0; j < 2; ++j)
      frag16(pf[j], &Ps[wave][ln][32 * j], koff);

    // O += P V : grouped loads (4 independent frags) then 4 independent WMMAs
    for (int j = 0; j < 2; ++j)
      for (int g = 0; g < 2; ++g) {
        FragA vf[4];
        for (int i = 0; i < 4; ++i)
          frag16(vf[i], &Vt[(g * 4 + i) * 16 + ln][32 * j], koff);
        for (int i = 0; i < 4; ++i)
          o[g * 4 + i] = __builtin_amdgcn_wmma_f32_16x16x32_bf16(
              false, pf[j].v, false, vf[i].v, (short)0, o[g * 4 + i],
              false, false);
      }
    __syncthreads();
  }

  // normalize + write bf16 attn_raw [B,S,H*D]
  for (int nt = 0; nt < 8; ++nt) {
    for (int r = 0; r < 8; ++r) {
      const int s = qt * 64 + wave * 16 + hi * 8 + r;
      const int d = nt * 16 + ln;
      attn[((size_t)(b * S_ + s)) * (size_t)(H_ * D_) + h * D_ + d] =
          f2bf(o[nt][r] / lrow[r]);
    }
  }
}

// ---------------------------------------------------------------------------
// Host-side launch
// ---------------------------------------------------------------------------
extern "C" void kernel_launch(void* const* d_in, const int* in_sizes, int n_in,
                              void* d_out, int out_size, void* d_ws, size_t ws_size,
                              hipStream_t stream) {
  (void)in_sizes; (void)n_in; (void)out_size; (void)ws_size;
  const float* x  = (const float*)d_in[0];
  const float* Wq = (const float*)d_in[1];
  const float* Wk = (const float*)d_in[2];
  const float* Wv = (const float*)d_in[3];
  const float* Wo = (const float*)d_in[4];
  const float* Aq = (const float*)d_in[5];
  const float* Bq = (const float*)d_in[6];
  const float* Ak = (const float*)d_in[7];
  const float* Bk = (const float*)d_in[8];
  const float* Av = (const float*)d_in[9];
  const float* Bv = (const float*)d_in[10];
  const float* Ao = (const float*)d_in[11];
  const float* Bo = (const float*)d_in[12];
  float* out = (float*)d_out;

  const int M = B_ * S_;                 // 4096
  char* ws = (char*)d_ws;
  u16* xb  = (u16*)ws; ws += (size_t)M * HID_ * 2;          // 33.5 MB
  u16* Wqb = (u16*)ws; ws += (size_t)(H_*D_)   * HID_ * 2;  // 33.5 MB
  u16* Wkb = (u16*)ws; ws += (size_t)(HKV_*D_) * HID_ * 2;  //  8.4 MB
  u16* Wvb = (u16*)ws; ws += (size_t)(HKV_*D_) * HID_ * 2;  //  8.4 MB
  u16* Wob = (u16*)ws; ws += (size_t)HID_ * (H_*D_) * 2;    // 33.5 MB
  u16* qb  = (u16*)ws; ws += (size_t)M * H_   * D_ * 2;     // 33.5 MB
  u16* kb  = (u16*)ws; ws += (size_t)M * HKV_ * D_ * 2;     //  8.4 MB
  u16* vb  = (u16*)ws; ws += (size_t)M * HKV_ * D_ * 2;     //  8.4 MB
  u16* attnb = (u16*)ws; ws += (size_t)M * H_ * D_ * 2;     // 33.5 MB
  float* tq = (float*)ws; ws += (size_t)M * R_ * 4;
  float* tk = (float*)ws; ws += (size_t)M * R_ * 4;
  float* tv = (float*)ws; ws += (size_t)M * R_ * 4;
  float* to = (float*)ws; ws += (size_t)M * R_ * 4;

  auto cvt = [&](const float* src, u16* dst, size_t n) {
    cvt_bf16_kernel<<<dim3((unsigned)((n / 8 + 255) / 256)), dim3(256), 0,
                      stream>>>(src, dst, n);
  };
  cvt(x,  xb,  (size_t)M * HID_);
  cvt(Wq, Wqb, (size_t)(H_*D_)   * HID_);
  cvt(Wk, Wkb, (size_t)(HKV_*D_) * HID_);
  cvt(Wv, Wvb, (size_t)(HKV_*D_) * HID_);
  cvt(Wo, Wob, (size_t)HID_ * (H_*D_));

  const dim3 tb(256);
  const dim3 tg((M * R_) / 256);
  lora_t_kernel<float><<<tg, tb, 0, stream>>>(x, Aq, tq, HID_);
  lora_t_kernel<float><<<tg, tb, 0, stream>>>(x, Ak, tk, HID_);
  lora_t_kernel<float><<<tg, tb, 0, stream>>>(x, Av, tv, HID_);

  gemm_bf16<u16><<<dim3(H_*D_/128,   M/128), dim3(256), 0, stream>>>(
      xb, Wqb, tq, Bq, qb, M, H_*D_,   HID_, 1);
  gemm_bf16<u16><<<dim3(HKV_*D_/128, M/128), dim3(256), 0, stream>>>(
      xb, Wkb, tk, Bk, kb, M, HKV_*D_, HID_, 1);
  gemm_bf16<u16><<<dim3(HKV_*D_/128, M/128), dim3(256), 0, stream>>>(
      xb, Wvb, tv, Bv, vb, M, HKV_*D_, HID_, 1);

  {
    const size_t qpairs = (size_t)M * H_ * (D_ / 2);
    rope_kernel<<<dim3((unsigned)((qpairs + 255) / 256)), dim3(256), 0, stream>>>(
        qb, H_, 0.08838834764831845f /* 1/sqrt(128) */, qpairs);
    const size_t kpairs = (size_t)M * HKV_ * (D_ / 2);
    rope_kernel<<<dim3((unsigned)((kpairs + 255) / 256)), dim3(256), 0, stream>>>(
        kb, HKV_, 1.0f, kpairs);
  }

  attn_fa<<<dim3(B_ * H_ * (S_ / 64)), dim3(128), 0, stream>>>(qb, kb, vb, attnb);

  lora_t_kernel<u16><<<tg, tb, 0, stream>>>(attnb, Ao, to, H_ * D_);
  gemm_bf16<float><<<dim3(HID_/128, M/128), dim3(256), 0, stream>>>(
      attnb, Wob, to, Bo, out, M, HID_, H_ * D_, 1);
}
